// Seq2seq_86105504350243
// MI455X (gfx1250) — matrix operand order repaired and checked by reading, not compiled, not run on hardware
//
#include <hip/hip_runtime.h>
#include <hip/hip_bf16.h>

#define BB 16
#define LL 512
#define TTK 128
#define VV 21128
#define DD 512

typedef __bf16 bf16_t;
typedef __attribute__((ext_vector_type(16))) __bf16 v16bf;
typedef __attribute__((ext_vector_type(8)))  __bf16 v8bf;
typedef __attribute__((ext_vector_type(8)))  float  v8f;

static __device__ __forceinline__ float sigmf(float x) { return 1.0f / (1.0f + __expf(-x)); }

static __device__ __forceinline__ v8f wmma_bf16(v16bf a, v16bf b, v8f c) {
  // D = A(16x32) * B(32x16) + C, f32 accumulate
  return __builtin_amdgcn_wmma_f32_16x16x32_bf16(false, a, false, b, (short)0, c, false, false);
}

// A fragment: 16x32 bf16, row-major source, lda in elements.
// lane<16: row=lane, k=k0..k0+15; lane>=16: row=lane-16, k=k0+16..k0+31.
static __device__ __forceinline__ v16bf load_a_frag(const bf16_t* A, int lda, int k0, int lane) {
  int hi = lane >> 4, r = lane & 15;
  const bf16_t* p = A + (size_t)r * lda + k0 + hi * 16;
  union { v16bf v; v8bf h[2]; } u;
  u.h[0] = *(const v8bf*)p;
  u.h[1] = *(const v8bf*)(p + 8);
  return u.v;
}

// B fragment: 32x16 where B[k][n] = W[n0+n][k0+k] (W row-major NxK). lane<16: n=lane, k=k0..k0+15.
// Out-of-range rows are CLAMPED (not zeroed): they only affect output columns that are never stored.
static __device__ __forceinline__ v16bf load_b_frag(const bf16_t* W, int ldw, int n0, int k0, int N, int lane) {
  int hi = lane >> 4, r = lane & 15;
  int row = n0 + r;
  row = (row < N) ? row : (N - 1);
  return *(const v16bf*)(W + (size_t)row * ldw + k0 + hi * 16);
}

// ------------------- elementwise kernels -------------------

__global__ void k_f2bf(const float* __restrict__ s, bf16_t* __restrict__ d, size_t n) {
  size_t i = (size_t)blockIdx.x * 256 + threadIdx.x;
  if (i < n) d[i] = (bf16_t)s[i];
}

__global__ void k_gather(const int* __restrict__ ids, const float* __restrict__ E,
                         bf16_t* __restrict__ dst, size_t total, int rows_per_b, int id_ld) {
  size_t i = (size_t)blockIdx.x * 256 + threadIdx.x;
  if (i >= total) return;
  int row = (int)(i >> 9);
  int d = (int)(i & 511);
  int b = row / rows_per_b;
  int rr = row - b * rows_per_b;
  int id = ids[b * id_ld + rr];
  dst[i] = (bf16_t)E[(size_t)id * DD + d];
}

__global__ void k_build_attnA(const bf16_t* __restrict__ hidden, const bf16_t* __restrict__ ef,
                              const bf16_t* __restrict__ eb, bf16_t* __restrict__ dst) {
  size_t total = (size_t)BB * LL * 1536;
  size_t i = (size_t)blockIdx.x * 256 + threadIdx.x;
  if (i >= total) return;
  int row = (int)(i / 1536);
  int col = (int)(i % 1536);
  int b = row >> 9;
  bf16_t v;
  if (col < 512)       v = hidden[b * 512 + col];
  else if (col < 1024) v = ef[(size_t)row * 512 + (col - 512)];
  else                 v = eb[(size_t)row * 512 + (col - 1024)];
  dst[i] = v;
}

__global__ void k_att_dot(const float* __restrict__ energy, const float* __restrict__ vw,
                          float* __restrict__ att) {
  int row = blockIdx.x;               // 0..8191
  __shared__ float red[64];
  float s = 0.f;
  for (int e = threadIdx.x; e < 512; e += 64) s += energy[(size_t)row * 512 + e] * vw[e];
  red[threadIdx.x] = s; __syncthreads();
  for (int off = 32; off > 0; off >>= 1) {
    if (threadIdx.x < off) red[threadIdx.x] += red[threadIdx.x + off];
    __syncthreads();
  }
  if (threadIdx.x == 0) att[row] = red[0];
}

__global__ void k_softmax(float* __restrict__ att, const int* __restrict__ doc_ids) {
  int b = blockIdx.x, l = threadIdx.x;      // 512 threads
  __shared__ float red[512];
  float v = att[b * 512 + l];
  if (doc_ids[b * 512 + l] == 0) v = -1e10f;
  red[l] = v; __syncthreads();
  for (int off = 256; off > 0; off >>= 1) {
    if (l < off) red[l] = fmaxf(red[l], red[l + off]);
    __syncthreads();
  }
  float mx = red[0]; __syncthreads();
  float e = __expf(v - mx);
  red[l] = e; __syncthreads();
  for (int off = 256; off > 0; off >>= 1) {
    if (l < off) red[l] += red[l + off];
    __syncthreads();
  }
  att[b * 512 + l] = e / red[0];
}

__global__ void k_weighted(const float* __restrict__ a, const bf16_t* __restrict__ ef,
                           const bf16_t* __restrict__ eb, bf16_t* __restrict__ w) {
  int b = blockIdx.x, e = threadIdx.x;      // 1024 threads
  const bf16_t* src = (e < 512) ? ef : eb;
  int d = e & 511;
  float s = 0.f;
  for (int l = 0; l < 512; ++l)
    s += a[b * 512 + l] * (float)src[((size_t)b * 512 + l) * 512 + d];
  w[b * 1024 + e] = (bf16_t)s;
}

__global__ void k_build_decx(const bf16_t* __restrict__ embs, const bf16_t* __restrict__ weighted,
                             bf16_t* __restrict__ dst) {
  size_t total = (size_t)2032 * 1536;
  size_t i = (size_t)blockIdx.x * 256 + threadIdx.x;
  if (i >= total) return;
  int row = (int)(i / 1536);
  int col = (int)(i % 1536);
  int b = row / 127;
  dst[i] = (col < 512) ? embs[(size_t)row * 512 + col] : weighted[b * 1024 + (col - 512)];
}

__global__ void k_dec_gates(const float* __restrict__ g, bf16_t* __restrict__ hdec) {
  size_t total = (size_t)2032 * 512;
  size_t i = (size_t)blockIdx.x * 256 + threadIdx.x;
  if (i >= total) return;
  int row = (int)(i >> 9);
  int d = (int)(i & 511);
  size_t base = (size_t)row * 2048;
  float iv = g[base + d];
  float gv = g[base + 1024 + d];
  float ov = g[base + 1536 + d];
  float c = sigmf(iv) * tanhf(gv);
  float h = sigmf(ov) * tanhf(c);
  hdec[i] = (bf16_t)h;
}

__global__ void k_build_predA(const bf16_t* __restrict__ hdec, const bf16_t* __restrict__ weighted,
                              const bf16_t* __restrict__ embs, bf16_t* __restrict__ dst) {
  size_t total = (size_t)2032 * 2048;
  size_t i = (size_t)blockIdx.x * 256 + threadIdx.x;
  if (i >= total) return;
  int row = (int)(i >> 11);
  int col = (int)(i & 2047);
  int b = row / 127;
  bf16_t v;
  if (col < 512)       v = hdec[(size_t)row * 512 + col];
  else if (col < 1536) v = weighted[b * 1024 + (col - 512)];
  else                 v = embs[(size_t)row * 512 + (col - 1536)];
  dst[i] = v;
}

// ------------------- generic WMMA GEMM: C(MxN) = A(MxK) * W(NxK)^T -------------------
// 256 threads = 8 waves; wave w owns 2 adjacent N-tiles (one A fragment feeds 2 WMMAs);
// block covers 256 output columns. M % 16 == 0.
__global__ void __launch_bounds__(256)
k_gemm(const bf16_t* __restrict__ A, const bf16_t* __restrict__ W,
       const float* __restrict__ bias1, const float* __restrict__ bias2,
       float* __restrict__ C, bf16_t* __restrict__ Cb,
       int M, int N, int K, int lda, int ldw, int ldc, int act) {
  int lane = threadIdx.x & 31;
  int wv = threadIdx.x >> 5;
  int n0 = (blockIdx.x * 16 + wv * 2) * 16;
  if (n0 >= N) return;
  int n1 = n0 + 16;
  int m0 = blockIdx.y * 16;
  const bf16_t* Arow = A + (size_t)m0 * lda;

  v8f acc0 = {}, acc1 = {};
  for (int k0 = 0; k0 < K; k0 += 32) {
    v16bf af = load_a_frag(Arow, lda, k0, lane);
    v16bf b0 = load_b_frag(W, ldw, n0, k0, N, lane);
    v16bf b1 = load_b_frag(W, ldw, n1, k0, N, lane);
    if (k0 + 64 < K) {
      int r = n0 + (lane & 15);
      r = (r < N) ? r : (N - 1);
      __builtin_prefetch((const void*)(W + (size_t)r * ldw + k0 + 64), 0, 1);
    }
    acc0 = wmma_bf16(af, b0, acc0);
    acc1 = wmma_bf16(af, b1, acc1);
  }

  int hi = lane >> 4, cl = lane & 15;
#pragma unroll
  for (int tile = 0; tile < 2; ++tile) {
    int col = (tile ? n1 : n0) + cl;
    if (col >= N) continue;
    const v8f& acc = tile ? acc1 : acc0;
    float bsum = (bias1 ? bias1[col] : 0.f) + (bias2 ? bias2[col] : 0.f);
#pragma unroll
    for (int e = 0; e < 8; ++e) {
      int row = m0 + e + hi * 8;
      float v = acc[e] + bsum;
      if (act == 1) v = tanhf(v);
      if (C)  C[(size_t)row * ldc + col] = v;
      if (Cb) Cb[(size_t)row * ldc + col] = (bf16_t)v;
    }
  }
}

// ------------------- persistent LSTM recurrence -------------------
// grid.x = 2 (direction). 1024 threads = 32 waves; wave w owns output columns [w*16, w*16+16)
// and gate rows {col, col+512, col+1024, col+1536} of Whh so i/f/g/o live in its registers.
__global__ void __launch_bounds__(1024)
k_lstm_rec(const float* __restrict__ xg_f, const float* __restrict__ xg_b,
           const bf16_t* __restrict__ Whh_f_bf, const bf16_t* __restrict__ Whh_b_bf,
           bf16_t* __restrict__ enc_f, bf16_t* __restrict__ enc_b,
           bf16_t* __restrict__ hcat) {
  const int dir = blockIdx.x;
  const float*  xg  = dir ? xg_b : xg_f;
  const bf16_t* Whh = dir ? Whh_b_bf : Whh_f_bf;
  bf16_t*       enc = dir ? enc_b : enc_f;

  __shared__ bf16_t h_s[16][520];   // padded rows: 1040B stride, 16B-aligned chunks
  int tid = threadIdx.x;
  int lane = tid & 31, wv = tid >> 5;
  int hi = lane >> 4, r = lane & 15;
  int col = wv * 16 + r;            // output column 0..511 for this lane's C slots

  for (int i = tid; i < 16 * 520; i += 1024) ((bf16_t*)h_s)[i] = (bf16_t)0.f;
  __syncthreads();

  float c[8];
#pragma unroll
  for (int e = 0; e < 8; ++e) c[e] = 0.f;

  const bf16_t* Wi = Whh + (size_t)(col) * 512;
  const bf16_t* Wf = Whh + (size_t)(512 + col) * 512;
  const bf16_t* Wg = Whh + (size_t)(1024 + col) * 512;
  const bf16_t* Wo = Whh + (size_t)(1536 + col) * 512;

  for (int s = 0; s < LL; ++s) {
    int t = dir ? (LL - 1 - s) : s;
    v8f ai = {}, af = {}, ag = {}, ao = {};
    for (int k0 = 0; k0 < 512; k0 += 32) {
      const bf16_t* hp = &h_s[r][k0 + hi * 16];
      union { v16bf v; v8bf h[2]; } ua;
      ua.h[0] = *(const v8bf*)hp;
      ua.h[1] = *(const v8bf*)(hp + 8);
      v16bf bi = *(const v16bf*)(Wi + k0 + hi * 16);
      v16bf bff = *(const v16bf*)(Wf + k0 + hi * 16);
      v16bf bg = *(const v16bf*)(Wg + k0 + hi * 16);
      v16bf bo = *(const v16bf*)(Wo + k0 + hi * 16);
      ai = wmma_bf16(ua.v, bi, ai);
      af = wmma_bf16(ua.v, bff, af);
      ag = wmma_bf16(ua.v, bg, ag);
      ao = wmma_bf16(ua.v, bo, ao);
    }
    __syncthreads();   // all reads of h_s (t-1) complete
#pragma unroll
    for (int e = 0; e < 8; ++e) {
      int b = e + hi * 8;
      const float* xrow = xg + ((size_t)b * LL + t) * 2048;
      float gi = ai[e] + xrow[col];
      float gf = af[e] + xrow[512 + col];
      float gg = ag[e] + xrow[1024 + col];
      float go = ao[e] + xrow[1536 + col];
      float iv = sigmf(gi), fv = sigmf(gf), gv = tanhf(gg), ov = sigmf(go);
      c[e] = fv * c[e] + iv * gv;
      float h = ov * tanhf(c[e]);
      h_s[b][col] = (bf16_t)h;
      enc[((size_t)b * LL + t) * 512 + col] = (bf16_t)h;
    }
    __syncthreads();   // h(t) visible before next step's reads
  }

  for (int i = tid; i < 16 * 512; i += 1024) {
    int b = i >> 9, d = i & 511;
    hcat[(size_t)b * 1024 + dir * 512 + d] = h_s[b][d];
  }
}

// ------------------- compact pred (B*(T-1), V) -> logits (B, V, T) via LDS transpose ----
// grid: (ceil(V/32), 4, B); block (32, 8). Both global sides fully coalesced.
// Also writes the t=0 zero plane (pred has no row for t=0).
__global__ void k_logits_tr(const float* __restrict__ predC, float* __restrict__ out) {
  __shared__ float tile[32][33];
  int b = blockIdx.z;
  int v0 = blockIdx.x * 32;
  int t0 = blockIdx.y * 32;
  int tx = threadIdx.x, ty = threadIdx.y;
#pragma unroll
  for (int j = 0; j < 4; ++j) {
    int tl = ty + j * 8;            // local t
    int t = t0 + tl;
    int v = v0 + tx;
    float val = 0.f;
    if (t >= 1 && v < VV)
      val = predC[((size_t)b * 127 + (t - 1)) * VV + v];
    tile[tl][tx] = val;
  }
  __syncthreads();
#pragma unroll
  for (int j = 0; j < 4; ++j) {
    int vl = ty + j * 8;            // local v
    int v = v0 + vl;
    int t = t0 + tx;
    if (v < VV)
      out[((size_t)b * VV + v) * TTK + t] = tile[tx][vl];
  }
}

// ------------------- NLL loss from compact pred -------------------
__global__ void k_loss(const float* __restrict__ predC, const int* __restrict__ sum_ids,
                       float* __restrict__ loss) {
  int rowid = blockIdx.x;                 // 0..B*T-1
  int b = rowid >> 7, t = rowid & 127;
  int tid = threadIdx.x;
  if (t == 0) {
    // logits[:, :, 0] == 0  ->  nll = log(V)
    if (tid == 0) atomicAdd(loss, __logf((float)VV) * (1.0f / (BB * TTK)));
    return;
  }
  const float* base = predC + ((size_t)b * 127 + (t - 1)) * VV;
  __shared__ float red[256];
  float mx = -3.4e38f;
  for (int v = tid; v < VV; v += 256) mx = fmaxf(mx, base[v]);
  red[tid] = mx; __syncthreads();
  for (int off = 128; off > 0; off >>= 1) {
    if (tid < off) red[tid] = fmaxf(red[tid], red[tid + off]);
    __syncthreads();
  }
  mx = red[0]; __syncthreads();
  float s = 0.f;
  for (int v = tid; v < VV; v += 256) s += __expf(base[v] - mx);
  red[tid] = s; __syncthreads();
  for (int off = 128; off > 0; off >>= 1) {
    if (tid < off) red[tid] += red[tid + off];
    __syncthreads();
  }
  if (tid == 0) {
    int tgt = sum_ids[b * TTK + t];
    float lp = base[tgt] - mx - __logf(red[0]);
    atomicAdd(loss, -lp * (1.0f / (BB * TTK)));
  }
}

// ------------------- launch -------------------

extern "C" void kernel_launch(void* const* d_in, const int* in_sizes, int n_in,
                              void* d_out, int out_size, void* d_ws, size_t ws_size,
                              hipStream_t stream) {
  const int*   doc_ids = (const int*)d_in[0];
  const int*   sum_ids = (const int*)d_in[1];
  const float* E       = (const float*)d_in[2];
  const float* Wih_f   = (const float*)d_in[3];
  const float* Whh_f   = (const float*)d_in[4];
  const float* bih_f   = (const float*)d_in[5];
  const float* bhh_f   = (const float*)d_in[6];
  const float* Wih_b   = (const float*)d_in[7];
  const float* Whh_b   = (const float*)d_in[8];
  const float* bih_b   = (const float*)d_in[9];
  const float* bhh_b   = (const float*)d_in[10];
  const float* fc_W    = (const float*)d_in[11];
  const float* fc_b    = (const float*)d_in[12];
  const float* attn_W  = (const float*)d_in[13];
  const float* attn_b  = (const float*)d_in[14];
  const float* v_w     = (const float*)d_in[15];
  const float* dWih    = (const float*)d_in[16];
  const float* dbih    = (const float*)d_in[17];
  const float* dbhh    = (const float*)d_in[18];
  const float* dfc_W   = (const float*)d_in[19];
  const float* dfc_b   = (const float*)d_in[20];
  float* out = (float*)d_out;

  char* cur = (char*)d_ws;
  auto alloc = [&](size_t bytes) -> char* {
    char* p = cur; cur += (bytes + 255) & ~(size_t)255; return p;
  };

  // bf16 weights (persistent)
  bf16_t* dfcW_bf  = (bf16_t*)alloc((size_t)VV * 2048 * 2);
  bf16_t* Wihf_bf  = (bf16_t*)alloc((size_t)2048 * 512 * 2);
  bf16_t* Wihb_bf  = (bf16_t*)alloc((size_t)2048 * 512 * 2);
  bf16_t* Whhf_bf  = (bf16_t*)alloc((size_t)2048 * 512 * 2);
  bf16_t* Whhb_bf  = (bf16_t*)alloc((size_t)2048 * 512 * 2);
  bf16_t* fcW_bf   = (bf16_t*)alloc((size_t)512 * 1024 * 2);
  bf16_t* attnW_bf = (bf16_t*)alloc((size_t)512 * 1536 * 2);
  bf16_t* dWih_bf  = (bf16_t*)alloc((size_t)2048 * 1536 * 2);
  // activations (persistent across phases)
  bf16_t* emb_bf   = (bf16_t*)alloc((size_t)BB * LL * 512 * 2);
  bf16_t* embs_bf  = (bf16_t*)alloc((size_t)2032 * 512 * 2);
  bf16_t* enc_f    = (bf16_t*)alloc((size_t)BB * LL * 512 * 2);
  bf16_t* enc_b    = (bf16_t*)alloc((size_t)BB * LL * 512 * 2);
  bf16_t* hcat     = (bf16_t*)alloc((size_t)BB * 1024 * 2);
  bf16_t* hidden   = (bf16_t*)alloc((size_t)BB * 512 * 2);
  float*  att      = (float*)alloc((size_t)BB * 512 * 4);
  bf16_t* weighted = (bf16_t*)alloc((size_t)BB * 1024 * 2);
  bf16_t* hdec     = (bf16_t*)alloc((size_t)2032 * 512 * 2);

  // big transient region with lifetime-based overlays:
  //   [0,128M):   xg_f/xg_b          (until recurrence done)
  //   [0, 42M):   attnA/energy       (attention phase)
  //   [44M,68M):  decx/g_dec         (decoder-gate phase)
  //   [0,164M):   predC              (written by pred GEMM, after all above are dead)
  //   [168M,176M): predA             (pred GEMM input; no overlap with predC)
  char* big = alloc((size_t)176 * 1024 * 1024);
  float*  xg_f   = (float*)big;                                        // 64 MB
  float*  xg_b   = (float*)(big + (size_t)64 * 1024 * 1024);           // 64 MB
  bf16_t* attnA  = (bf16_t*)big;                                       // 24 MB
  float*  energy = (float*)(big + (size_t)26 * 1024 * 1024);           // 16 MB
  bf16_t* decx   = (bf16_t*)(big + (size_t)44 * 1024 * 1024);          // 6 MB
  float*  g_dec  = (float*)(big + (size_t)52 * 1024 * 1024);           // 16 MB
  float*  predC  = (float*)big;                                        // 164 MB
  bf16_t* predA  = (bf16_t*)(big + (size_t)168 * 1024 * 1024);         // 8 MB

  auto blocks = [](size_t n) { return (unsigned)((n + 255) / 256); };

  // ---- weight conversion f32 -> bf16 ----
  k_f2bf<<<blocks((size_t)VV * 2048), 256, 0, stream>>>(dfc_W, dfcW_bf, (size_t)VV * 2048);
  k_f2bf<<<blocks((size_t)2048 * 512), 256, 0, stream>>>(Wih_f, Wihf_bf, (size_t)2048 * 512);
  k_f2bf<<<blocks((size_t)2048 * 512), 256, 0, stream>>>(Wih_b, Wihb_bf, (size_t)2048 * 512);
  k_f2bf<<<blocks((size_t)2048 * 512), 256, 0, stream>>>(Whh_f, Whhf_bf, (size_t)2048 * 512);
  k_f2bf<<<blocks((size_t)2048 * 512), 256, 0, stream>>>(Whh_b, Whhb_bf, (size_t)2048 * 512);
  k_f2bf<<<blocks((size_t)512 * 1024), 256, 0, stream>>>(fc_W, fcW_bf, (size_t)512 * 1024);
  k_f2bf<<<blocks((size_t)512 * 1536), 256, 0, stream>>>(attn_W, attnW_bf, (size_t)512 * 1536);
  k_f2bf<<<blocks((size_t)2048 * 1536), 256, 0, stream>>>(dWih, dWih_bf, (size_t)2048 * 1536);

  // ---- embedding gathers ----
  k_gather<<<blocks((size_t)BB * LL * 512), 256, 0, stream>>>(doc_ids, E, emb_bf,
                                                              (size_t)BB * LL * 512, 512, 512);
  k_gather<<<blocks((size_t)2032 * 512), 256, 0, stream>>>(sum_ids, E, embs_bf,
                                                           (size_t)2032 * 512, 127, 128);

  auto gemm = [&](const bf16_t* A, const bf16_t* W, const float* b1, const float* b2,
                  float* C, bf16_t* Cb, int M, int N, int K, int lda, int ldw, int ldc,
                  int act) {
    dim3 grid((N + 255) / 256, M / 16);
    k_gemm<<<grid, 256, 0, stream>>>(A, W, b1, b2, C, Cb, M, N, K, lda, ldw, ldc, act);
  };

  // ---- encoder input-gate GEMMs: xg = emb @ Wih^T + bih + bhh ----
  gemm(emb_bf, Wihf_bf, bih_f, bhh_f, xg_f, nullptr, BB * LL, 2048, 512, 512, 512, 2048, 0);
  gemm(emb_bf, Wihb_bf, bih_b, bhh_b, xg_b, nullptr, BB * LL, 2048, 512, 512, 512, 2048, 0);

  // ---- bidirectional LSTM recurrence (1 WGP per direction) ----
  k_lstm_rec<<<dim3(2), dim3(1024), 0, stream>>>(xg_f, xg_b, Whhf_bf, Whhb_bf,
                                                 enc_f, enc_b, hcat);

  // ---- hidden = tanh(hcat @ fc_W^T + fc_b) ----
  gemm(hcat, fcW_bf, fc_b, nullptr, nullptr, hidden, BB, 512, 1024, 1024, 1024, 512, 1);

  // ---- attention ----
  k_build_attnA<<<blocks((size_t)BB * LL * 1536), 256, 0, stream>>>(hidden, enc_f, enc_b, attnA);
  gemm(attnA, attnW_bf, attn_b, nullptr, energy, nullptr, BB * LL, 512, 1536, 1536, 1536, 512, 1);
  k_att_dot<<<BB * LL, 64, 0, stream>>>(energy, v_w, att);
  k_softmax<<<BB, 512, 0, stream>>>(att, doc_ids);
  k_weighted<<<BB, 1024, 0, stream>>>(att, enc_f, enc_b, weighted);

  // ---- decoder gates ----
  k_build_decx<<<blocks((size_t)2032 * 1536), 256, 0, stream>>>(embs_bf, weighted, decx);
  gemm(decx, dWih_bf, dbih, dbhh, g_dec, nullptr, 2032, 2048, 1536, 1536, 1536, 2048, 0);
  k_dec_gates<<<blocks((size_t)2032 * 512), 256, 0, stream>>>(g_dec, hdec);

  // ---- vocab projection (compact, coalesced) then LDS-tiled transpose into logits ----
  k_build_predA<<<blocks((size_t)2032 * 2048), 256, 0, stream>>>(hdec, weighted, embs_bf, predA);
  gemm(predA, dfcW_bf, dfc_b, nullptr, predC, nullptr, 2032, VV, 2048, 2048, 2048, VV, 0);
  k_logits_tr<<<dim3((VV + 31) / 32, 4, BB), dim3(32, 8), 0, stream>>>(predC, out);

  // ---- NLL loss (zero only the scalar slot) ----
  hipMemsetAsync(out + (size_t)BB * VV * TTK, 0, sizeof(float), stream);
  k_loss<<<BB * TTK, 256, 0, stream>>>(predC, sum_ids, out + (size_t)BB * VV * TTK);
}
